// MultiHeadSelfAttention_27745488732283
// MI455X (gfx1250) — matrix operand order repaired and checked by reading, not compiled, not run on hardware
//
#include <hip/hip_runtime.h>

// ---------------- problem constants ----------------
#define D_MODEL   512
#define N_HEADS   16
#define HEAD_DIM  32          // D_MODEL / N_HEADS
#define BATCH     2
#define SEQ       2048
#define BT        (BATCH*SEQ)       // 4096 tokens total
#define QKV_N     (3*D_MODEL)       // 1536
#define NEG_BIG   (-1.0e30f)

typedef __attribute__((ext_vector_type(16))) _Float16 v16h;
typedef __attribute__((ext_vector_type(8)))  float    v8f;

static __device__ __forceinline__ v8f wmma_f16(v16h a, v16h b, v8f c) {
  // D = A(16x32 f16) * B(32x16 f16) + C(16x16 f32)
  return __builtin_amdgcn_wmma_f32_16x16x32_f16(
      /*neg_a=*/false, a, /*neg_b=*/false, b,
      /*c_mod=*/(short)0, c, /*reuse_a=*/false, /*reuse_b=*/false);
}

// A fragment: 16x32 f16 tile, row-major source with leading dim lda.
// Code's own k order: lane m=lane&15 holds row m; lane group kh=lane>>4 selects
// K chunk {kh*16 .. kh*16+15} (contiguous, matching the B fills).
static __device__ __forceinline__ v16h load_a_frag(const _Float16* __restrict__ A,
                                                   int lda, int lane) {
  int m  = lane & 15;
  int kh = lane >> 4;
  const _Float16* row = A + m * lda + kh * 16;
  v16h a;
#pragma unroll
  for (int i = 0; i < 16; ++i) a[i] = row[i];
  return a;
}

// ---------------- 1) f32 -> f16 conversion ----------------
__global__ void cvt_f32_f16(const float* __restrict__ src,
                            _Float16* __restrict__ dst, int n) {
  int i = blockIdx.x * blockDim.x + threadIdx.x;
  if (i < n) dst[i] = (_Float16)src[i];
}

// ---------------- 2) QKV projection (32x64 tile per wave) ----------------
// qkv[g][c] = sum_k x[g][k] * w_qkv[c][k];   c = which*512 + head*32 + d
// q,k scattered as (B*H, T, hd); v scattered TRANSPOSED as (B*H, hd, T)
// so the attention kernel's V B-fragments are contiguous loads.
__global__ void qkv_gemm(const _Float16* __restrict__ xh,
                         const _Float16* __restrict__ wh,
                         _Float16* __restrict__ qb,
                         _Float16* __restrict__ kb,
                         _Float16* __restrict__ vtb) {
  const int MT = BT / 32;       // 128
  const int NT = QKV_N / 64;    // 24
  int wave = threadIdx.x >> 5;
  int lane = threadIdx.x & 31;
  int w = blockIdx.x * (blockDim.x >> 5) + wave;
  if (w >= MT * NT) return;
  int mt = w % MT;
  int nt = w / MT;

  int nl = lane & 15;
  int kh = lane >> 4;

  v8f acc[2][4];
#pragma unroll
  for (int mi_ = 0; mi_ < 2; ++mi_)
#pragma unroll
    for (int j = 0; j < 4; ++j) acc[mi_][j] = (v8f){};

  for (int k0 = 0; k0 < D_MODEL; k0 += 32) {
    v16h a0 = load_a_frag(xh + (size_t)(mt * 32)      * D_MODEL + k0, D_MODEL, lane);
    v16h a1 = load_a_frag(xh + (size_t)(mt * 32 + 16) * D_MODEL + k0, D_MODEL, lane);
#pragma unroll
    for (int j = 0; j < 4; ++j) {
      const _Float16* wrow =
          wh + (size_t)(nt * 64 + j * 16 + nl) * D_MODEL + k0 + kh * 16;
      v16h b;
#pragma unroll
      for (int i = 0; i < 16; ++i) b[i] = wrow[i];
      acc[0][j] = wmma_f16(a0, b, acc[0][j]);
      acc[1][j] = wmma_f16(a1, b, acc[1][j]);
    }
  }

#pragma unroll
  for (int j = 0; j < 4; ++j) {
    int ncol  = nt * 64 + j * 16 + nl;
    int which = ncol >> 9;          // /512  (uniform within a j-group)
    int rem   = ncol & 511;
    int head  = rem >> 5;
    int d     = rem & 31;
#pragma unroll
    for (int mi_ = 0; mi_ < 2; ++mi_) {
#pragma unroll
      for (int r = 0; r < 8; ++r) {
        int g  = mt * 32 + mi_ * 16 + r + 8 * kh;   // global token row
        int bb = g >> 11;                           // /SEQ
        int t  = g & (SEQ - 1);
        _Float16 val = (_Float16)acc[mi_][j][r];
        if (which == 2) {
          // V transposed: (B*H, hd, T)
          vtb[(((size_t)bb * N_HEADS + head) * HEAD_DIM + d) * SEQ + t] = val;
        } else {
          _Float16* dst = (which == 0) ? qb : kb;
          dst[(((size_t)bb * N_HEADS + head) * SEQ + t) * HEAD_DIM + d] = val;
        }
      }
    }
  }
}

// ---------------- 3) RoPE (in place, f32 math; q,k only) ----------------
__global__ void rope_kernel(_Float16* __restrict__ qb, _Float16* __restrict__ kb) {
  int idx = blockIdx.x * blockDim.x + threadIdx.x;
  const int total = BATCH * N_HEADS * SEQ * 16;
  if (idx >= total) return;
  int j  = idx & 15;
  int t  = (idx >> 4) & (SEQ - 1);
  int bh = idx >> 15;
  size_t base = ((size_t)bh * SEQ + t) * HEAD_DIM;
  // inv_freq = base^(-2j/32) = exp(-ln(10000)*j/16)
  float inv_freq = expf(-(float)j * (9.210340371976184f / 16.0f));
  float ang = (float)t * inv_freq;
  float cs = cosf(ang), sn = sinf(ang);

  float q1 = (float)qb[base + j],      q2 = (float)qb[base + 16 + j];
  qb[base + j]      = (_Float16)(q1 * cs - q2 * sn);
  qb[base + 16 + j] = (_Float16)(q2 * cs + q1 * sn);

  float k1 = (float)kb[base + j],      k2 = (float)kb[base + 16 + j];
  kb[base + j]      = (_Float16)(k1 * cs - k2 * sn);
  kb[base + 16 + j] = (_Float16)(k2 * cs + k1 * sn);
}

// ---------------- 4) flash attention (64-key blocks, V^T layout) ----------------
// One wave = one (b,h) x 16 query rows. Per 64-key block:
// 4 S-WMMAs (K dim = hd = 32), one online-softmax pass, one LDS P
// round-trip (C-frag -> A-frag), 4 O-WMMAs. All fragment loads contiguous.
__global__ void flash_attn(const _Float16* __restrict__ qb,
                           const _Float16* __restrict__ kb,
                           const _Float16* __restrict__ vtb,
                           _Float16* __restrict__ ctx) {
  __shared__ _Float16 plds[4][16 * 66];   // per-wave 16x64 P tile, stride 66
  const int QT = SEQ / 16;                // 128 query tiles
  int wave = threadIdx.x >> 5;
  int lane = threadIdx.x & 31;
  int w = blockIdx.x * 4 + wave;
  if (w >= BATCH * N_HEADS * QT) return;
  int qt = (QT - 1) - (w % QT);           // long (late) tiles launch first
  int bh = w / QT;

  const _Float16* Q  = qb  + ((size_t)bh * SEQ + qt * 16) * HEAD_DIM;
  const _Float16* K  = kb  + (size_t)bh * SEQ * HEAD_DIM;
  const _Float16* Vt = vtb + (size_t)bh * HEAD_DIM * SEQ;   // (hd, T)
  _Float16* P = plds[wave];

  int nl = lane & 15;
  int kh = lane >> 4;
  const float scale = 0.17677669529663687f;   // 1/sqrt(32)

  v16h aq = load_a_frag(Q, HEAD_DIM, lane);   // Q fragment, reused every block

  float mi[8], li[8];
  v8f o0 = {}, o1 = {};
#pragma unroll
  for (int r = 0; r < 8; ++r) { mi[r] = NEG_BIG; li[r] = 0.0f; }

  int q0 = qt * 16;
  int nb = ((q0 + 15) >> 6) + 1;              // number of 64-key blocks (causal)

  for (int blk = 0; blk < nb; ++blk) {
    int tok0 = blk * 64;

    // ---- S = Q @ K^T  (four 16-col tiles) ----
    v8f s[4];
#pragma unroll
    for (int j = 0; j < 4; ++j) {
      const _Float16* kp = K + (size_t)(tok0 + j * 16 + nl) * HEAD_DIM + kh * 16;
      v16h bk;
#pragma unroll
      for (int i = 0; i < 16; ++i) bk[i] = kp[i];
      v8f z = {};
      s[j] = wmma_f16(aq, bk, z);
    }

    // ---- causal mask + scale; row max over all 4 tiles ----
    float rowm[8];
#pragma unroll
    for (int r = 0; r < 8; ++r) rowm[r] = NEG_BIG;
#pragma unroll
    for (int j = 0; j < 4; ++j) {
#pragma unroll
      for (int r = 0; r < 8; ++r) {
        int ki = tok0 + j * 16 + nl;
        int qi = q0 + r + 8 * kh;
        float v = (ki <= qi) ? s[j][r] * scale : NEG_BIG;
        s[j][r] = v;
        rowm[r] = fmaxf(rowm[r], v);
      }
    }
#pragma unroll
    for (int off = 1; off < 16; off <<= 1) {
#pragma unroll
      for (int r = 0; r < 8; ++r)
        rowm[r] = fmaxf(rowm[r], __shfl_xor(rowm[r], off, 32));
    }

    // ---- online rescale (once per 64 keys) ----
    float rs[8];
#pragma unroll
    for (int r = 0; r < 8; ++r) {
      float mnew = fmaxf(mi[r], rowm[r]);
      float corr = __expf(mi[r] - mnew);
      mi[r] = mnew;
      li[r] *= corr;
      o0[r] *= corr;
      o1[r] *= corr;
      rs[r] = 0.0f;
    }
#pragma unroll
    for (int j = 0; j < 4; ++j) {
#pragma unroll
      for (int r = 0; r < 8; ++r) {
        float p = __expf(s[j][r] - mi[r]);
        s[j][r] = p;
        rs[r] += p;
      }
    }
#pragma unroll
    for (int off = 1; off < 16; off <<= 1) {
#pragma unroll
      for (int r = 0; r < 8; ++r) rs[r] += __shfl_xor(rs[r], off, 32);
    }
#pragma unroll
    for (int r = 0; r < 8; ++r) li[r] += rs[r];

    // ---- C-fragment P -> LDS -> A-fragments (per-wave region, in-order DS) ----
#pragma unroll
    for (int j = 0; j < 4; ++j) {
#pragma unroll
      for (int r = 0; r < 8; ++r)
        P[(r + 8 * kh) * 66 + j * 16 + nl] = (_Float16)s[j][r];
    }
    v16h ap0 = load_a_frag(P, 66, lane);        // K cols  0..31 of P
    v16h ap1 = load_a_frag(P + 32, 66, lane);   // K cols 32..63 of P

    // ---- O += P @ V  (64 keys x 32 dims); V^T makes these contiguous ----
#pragma unroll
    for (int half = 0; half < 2; ++half) {
      const _Float16* vp0 = Vt + (size_t)nl        * SEQ + tok0 + half * 32 + kh * 16;
      const _Float16* vp1 = Vt + (size_t)(16 + nl) * SEQ + tok0 + half * 32 + kh * 16;
      v16h bv0, bv1;
#pragma unroll
      for (int i = 0; i < 16; ++i) { bv0[i] = vp0[i]; bv1[i] = vp1[i]; }
      v16h ap = half ? ap1 : ap0;
      o0 = wmma_f16(ap, bv0, o0);
      o1 = wmma_f16(ap, bv1, o1);
    }
  }

  // ---- normalize, store context as (B*T, D_MODEL) f16 ----
  int b_ = bh >> 4;
  int h  = bh & 15;
#pragma unroll
  for (int r = 0; r < 8; ++r) {
    int qi = q0 + r + 8 * kh;
    size_t g = (size_t)b_ * SEQ + qi;
    float inv = 1.0f / li[r];
    ctx[g * D_MODEL + h * HEAD_DIM + nl]      = (_Float16)(o0[r] * inv);
    ctx[g * D_MODEL + h * HEAD_DIM + 16 + nl] = (_Float16)(o1[r] * inv);
  }
}

// ---------------- 5) output projection (32x64 tile per wave, fp32 out) ----------------
__global__ void out_gemm(const _Float16* __restrict__ ctx,
                         const _Float16* __restrict__ woh,
                         float* __restrict__ out) {
  const int MT = BT / 32;        // 128
  const int NT = D_MODEL / 64;   // 8
  int wave = threadIdx.x >> 5;
  int lane = threadIdx.x & 31;
  int w = blockIdx.x * (blockDim.x >> 5) + wave;
  if (w >= MT * NT) return;
  int mt = w % MT;
  int nt = w / MT;

  int nl = lane & 15;
  int kh = lane >> 4;

  v8f acc[2][4];
#pragma unroll
  for (int mi_ = 0; mi_ < 2; ++mi_)
#pragma unroll
    for (int j = 0; j < 4; ++j) acc[mi_][j] = (v8f){};

  for (int k0 = 0; k0 < D_MODEL; k0 += 32) {
    v16h a0 = load_a_frag(ctx + (size_t)(mt * 32)      * D_MODEL + k0, D_MODEL, lane);
    v16h a1 = load_a_frag(ctx + (size_t)(mt * 32 + 16) * D_MODEL + k0, D_MODEL, lane);
#pragma unroll
    for (int j = 0; j < 4; ++j) {
      const _Float16* wrow =
          woh + (size_t)(nt * 64 + j * 16 + nl) * D_MODEL + k0 + kh * 16;
      v16h b;
#pragma unroll
      for (int i = 0; i < 16; ++i) b[i] = wrow[i];
      acc[0][j] = wmma_f16(a0, b, acc[0][j]);
      acc[1][j] = wmma_f16(a1, b, acc[1][j]);
    }
  }

#pragma unroll
  for (int j = 0; j < 4; ++j) {
    int ncol = nt * 64 + j * 16 + nl;
#pragma unroll
    for (int mi_ = 0; mi_ < 2; ++mi_) {
#pragma unroll
      for (int r = 0; r < 8; ++r) {
        int g = mt * 32 + mi_ * 16 + r + 8 * kh;
        out[(size_t)g * D_MODEL + ncol] = acc[mi_][j][r];
      }
    }
  }
}

// ---------------- launch ----------------
extern "C" void kernel_launch(void* const* d_in, const int* in_sizes, int n_in,
                              void* d_out, int out_size, void* d_ws, size_t ws_size,
                              hipStream_t stream) {
  const float* x    = (const float*)d_in[0];   // (B,T,512)
  const float* wqkv = (const float*)d_in[1];   // (1536,512)
  const float* wo   = (const float*)d_in[2];   // (512,512)
  float* out = (float*)d_out;

  _Float16* ws = (_Float16*)d_ws;
  size_t off = 0;
  _Float16* xh    = ws + off; off += (size_t)BT * D_MODEL;
  _Float16* wqkvh = ws + off; off += (size_t)QKV_N * D_MODEL;
  _Float16* woh   = ws + off; off += (size_t)D_MODEL * D_MODEL;
  _Float16* qh    = ws + off; off += (size_t)BATCH * N_HEADS * SEQ * HEAD_DIM;
  _Float16* kh    = ws + off; off += (size_t)BATCH * N_HEADS * SEQ * HEAD_DIM;
  _Float16* vth   = ws + off; off += (size_t)BATCH * N_HEADS * SEQ * HEAD_DIM;
  _Float16* ctx   = ws + off; off += (size_t)BT * D_MODEL;

  int nx = BT * D_MODEL, nw1 = QKV_N * D_MODEL, nw2 = D_MODEL * D_MODEL;
  cvt_f32_f16<<<(nx  + 255) / 256, 256, 0, stream>>>(x,    xh,    nx);
  cvt_f32_f16<<<(nw1 + 255) / 256, 256, 0, stream>>>(wqkv, wqkvh, nw1);
  cvt_f32_f16<<<(nw2 + 255) / 256, 256, 0, stream>>>(wo,   woh,   nw2);

  {
    int waves = (BT / 32) * (QKV_N / 64);      // 3072
    qkv_gemm<<<waves / 4, 128, 0, stream>>>(xh, wqkvh, qh, kh, vth);
  }
  {
    int total = BATCH * N_HEADS * SEQ * 16;    // 1,048,576
    rope_kernel<<<(total + 255) / 256, 256, 0, stream>>>(qh, kh);
  }
  {
    int waves = BATCH * N_HEADS * (SEQ / 16);  // 4096
    flash_attn<<<waves / 4, 128, 0, stream>>>(qh, kh, vth, ctx);
  }
  {
    int waves = (BT / 32) * (D_MODEL / 64);    // 1024
    out_gemm<<<waves / 4, 128, 0, stream>>>(ctx, woh, out);
  }
}
